// TransformerBlock_30176440222440
// MI455X (gfx1250) — compile-verified, matrix operand order checked by reading, not run on hardware
//
#include <hip/hip_runtime.h>

// ---------------- types ----------------
typedef __bf16 bf16_t;
typedef __attribute__((ext_vector_type(16))) __bf16 v16bf;
typedef __attribute__((ext_vector_type(8)))  float  v8f;
typedef __attribute__((ext_vector_type(4)))  unsigned int u32x4;

#define WMMA_BF16(a, b, c) \
  __builtin_amdgcn_wmma_f32_16x16x32_bf16(false, (a), false, (b), (short)0, (c), false, false)

// Load a 16x32 bf16 A/B fragment from a row-major (row = lane&15) buffer.
// Per ISA 16-bit layout: lanes 0-15 hold K [0..8)+[16..24), lanes 16-31 hold
// K [8..16)+[24..32). p = row base already offset to the 32-wide K chunk.
__device__ __forceinline__ v16bf frag_ld(const bf16_t* p, int halfw) {
  union { v16bf v; u32x4 q[2]; } u;
  u.q[0] = *(const u32x4*)(p + halfw * 8);
  u.q[1] = *(const u32x4*)(p + 16 + halfw * 8);
  return u.v;
}

// ---------------- cast fp32 -> bf16 ----------------
__global__ void cast_bf16_kernel(const float* __restrict__ in, bf16_t* __restrict__ out, long n) {
  long i = (long)blockIdx.x * blockDim.x + threadIdx.x;
  long stride = (long)gridDim.x * blockDim.x;
  for (; i < n; i += stride) out[i] = (bf16_t)in[i];
}

// ---------------- RMSNorm: f32 in -> bf16 out ----------------
__global__ __launch_bounds__(256) void rmsnorm_kernel(
    const float* __restrict__ x, const float* __restrict__ g,
    bf16_t* __restrict__ out, int D) {
  const int row = blockIdx.x;
  const float* xr = x + (size_t)row * D;
  bf16_t* orow = out + (size_t)row * D;
  float s = 0.f;
  for (int i = threadIdx.x; i < D; i += blockDim.x) { float v = xr[i]; s += v * v; }
#pragma unroll
  for (int m = 16; m >= 1; m >>= 1) s += __shfl_xor(s, m, 32);
  __shared__ float red[8];
  const int wid = threadIdx.x >> 5, lid = threadIdx.x & 31;
  if (lid == 0) red[wid] = s;
  __syncthreads();
  if (wid == 0) {
    float t = (lid < 8) ? red[lid] : 0.f;
#pragma unroll
    for (int m = 4; m >= 1; m >>= 1) t += __shfl_xor(t, m, 32);
    if (lid == 0) red[0] = t;
  }
  __syncthreads();
  const float inv = rsqrtf(red[0] / (float)D + 1e-5f);
  for (int i = threadIdx.x; i < D; i += blockDim.x)
    orow[i] = (bf16_t)(xr[i] * inv * g[i]);
}

// ---------------- NT GEMM: C[M,N] = A[M,K] * W[N,K]^T (bf16 -> f32 acc) ----
// Block tile 256x128, 8 waves as 4(M) x 2(N), 64x64 per wave = 16 WMMAs/K-step.
// Register double-buffered global->LDS staging (loads overlap WMMA compute).
#define GBM 256
#define GBN 128
#define GBK 32
#define GLD 40   // padded LDS row stride (elements); 80B = 5*16B aligned

__global__ __launch_bounds__(256) void gemm_bf16_nt_kernel(
    const bf16_t* __restrict__ A, const bf16_t* __restrict__ W,
    float* __restrict__ Cf, bf16_t* __restrict__ Cb,
    const float* __restrict__ resid, int M, int N, int K) {
  __shared__ bf16_t As[GBM * GLD];   // 20 KB
  __shared__ bf16_t Bs[GBN * GLD];   // 10 KB
  const int tid = threadIdx.x;
  const int m0 = blockIdx.y * GBM, n0 = blockIdx.x * GBN;
  const int wid = tid >> 5, lane = tid & 31;
  const int wm = wid & 3, wn = wid >> 2;     // 4 x 2 wave grid -> 64x64 per wave
  const int halfw = lane >> 4, nl = lane & 15;

  // staging geometry: chunk c covers row lin>>2, cols (lin&3)*8 .. +8
  const int sr = tid >> 2, sc = (tid & 3) << 3;

  v8f acc[4][4];
  {
    v8f z = {};
#pragma unroll
    for (int i = 0; i < 4; ++i)
#pragma unroll
      for (int j = 0; j < 4; ++j) acc[i][j] = z;
  }

  u32x4 ra[4], rb[2];
  // prologue: fetch K-step 0 into registers
#pragma unroll
  for (int c = 0; c < 4; ++c)
    ra[c] = *(const u32x4*)(A + (size_t)(m0 + sr + c * 64) * K + sc);
#pragma unroll
  for (int c = 0; c < 2; ++c)
    rb[c] = *(const u32x4*)(W + (size_t)(n0 + sr + c * 64) * K + sc);

  for (int k0 = 0; k0 < K; k0 += GBK) {
    // commit staged registers to LDS
#pragma unroll
    for (int c = 0; c < 4; ++c)
      *(u32x4*)(As + (sr + c * 64) * GLD + sc) = ra[c];
#pragma unroll
    for (int c = 0; c < 2; ++c)
      *(u32x4*)(Bs + (sr + c * 64) * GLD + sc) = rb[c];
    __syncthreads();

    // issue next K-step's global loads early; they retire during the WMMAs
    if (k0 + GBK < K) {
      const int kn = k0 + GBK;
#pragma unroll
      for (int c = 0; c < 4; ++c)
        ra[c] = *(const u32x4*)(A + (size_t)(m0 + sr + c * 64) * K + kn + sc);
#pragma unroll
      for (int c = 0; c < 2; ++c)
        rb[c] = *(const u32x4*)(W + (size_t)(n0 + sr + c * 64) * K + kn + sc);
    }

    v16bf bfr[4];
#pragma unroll
    for (int nt = 0; nt < 4; ++nt)
      bfr[nt] = frag_ld(Bs + (wn * 64 + nt * 16 + nl) * GLD, halfw);
#pragma unroll
    for (int mt = 0; mt < 4; ++mt) {
      v16bf afr = frag_ld(As + (wm * 64 + mt * 16 + nl) * GLD, halfw);
#pragma unroll
      for (int nt = 0; nt < 4; ++nt)
        acc[mt][nt] = WMMA_BF16(afr, bfr[nt], acc[mt][nt]);
    }
    __syncthreads();
  }

  // epilogue: C layout -> row = j + 8*halfw, col = nl
#pragma unroll
  for (int mt = 0; mt < 4; ++mt)
#pragma unroll
    for (int nt = 0; nt < 4; ++nt) {
      const int r0 = m0 + wm * 64 + mt * 16 + halfw * 8;
      const int cc = n0 + wn * 64 + nt * 16 + nl;
#pragma unroll
      for (int j = 0; j < 8; ++j) {
        const size_t off = (size_t)(r0 + j) * N + cc;
        const float v = acc[mt][nt][j];
        if (Cb) Cb[off] = (bf16_t)v;
        else    Cf[off] = resid ? (v + resid[off]) : v;
      }
    }
}

// ---------------- RoPE in-place on bf16 [B,S,H,Dh] ----------------
__global__ void rope_kernel(bf16_t* __restrict__ t, int S, int H, long total) {
  long i = (long)blockIdx.x * blockDim.x + threadIdx.x;
  if (i >= total) return;
  const int Dh = 64, hp = 32;
  const int p = (int)(i % hp);
  const long r = i / hp;            // (b*S + s)*H + h
  const int h = (int)(r % H);
  const long bs = r / H;
  const int s = (int)(bs % S);
  bf16_t* base = t + (size_t)bs * (H * Dh) + h * Dh;
  const float freq = __expf(-9.21034037f * (float)p / 32.f);  // theta^(-2p/Dh)
  const float ang = (float)s * freq;
  const float c = __cosf(ang), sn = __sinf(ang);
  const float x1 = (float)base[2 * p], x2 = (float)base[2 * p + 1];
  base[2 * p]     = (bf16_t)(x1 * c - x2 * sn);
  base[2 * p + 1] = (bf16_t)(x1 * sn + x2 * c);
}

// ---------------- causal flash attention, Dh=64, bf16 in/out -------------
__global__ __launch_bounds__(128) void attn_kernel(
    const bf16_t* __restrict__ Q, const bf16_t* __restrict__ Kmat,
    const bf16_t* __restrict__ V, bf16_t* __restrict__ O, int S, int H) {
  const int Dh = 64, LD = 72;          // 144B rows, 16B aligned
  __shared__ bf16_t Ks[64 * LD];       // K tile, row = key, col = d
  __shared__ bf16_t Vt[64 * LD];       // transposed V tile, row = d, col = key
  __shared__ bf16_t Pb[4][16 * LD];    // per-wave P buffer 16x64

  const int qt = blockIdx.x;           // 64-query tile
  const int bh = blockIdx.y;
  const int b = bh / H, h = bh % H;
  const int tid = threadIdx.x, wid = tid >> 5, lane = tid & 31;
  const int halfw = lane >> 4, nl = lane & 15;
  const size_t ld = (size_t)H * Dh;    // == D

  const bf16_t* Qb = Q + (size_t)b * S * ld + (size_t)h * Dh;
  const bf16_t* Kb = Kmat + (size_t)b * S * ld + (size_t)h * Dh;
  const bf16_t* Vb = V + (size_t)b * S * ld + (size_t)h * Dh;
  bf16_t* Ob = O + (size_t)b * S * ld + (size_t)h * Dh;

  // Q fragments for this wave's 16 rows, loaded directly in A layout
  const int qrow = qt * 64 + wid * 16 + nl;
  v16bf qf[2];
#pragma unroll
  for (int ks = 0; ks < 2; ++ks) {
    union { v16bf v; u32x4 q[2]; } u;
    const bf16_t* p = Qb + (size_t)qrow * ld + ks * 32;
    u.q[0] = *(const u32x4*)(p + halfw * 8);
    u.q[1] = *(const u32x4*)(p + 16 + halfw * 8);
    qf[ks] = u.v;
  }

  v8f oacc[4];
  {
    v8f z = {};
#pragma unroll
    for (int d = 0; d < 4; ++d) oacc[d] = z;
  }
  float mrow[8], lrow[8];
#pragma unroll
  for (int j = 0; j < 8; ++j) { mrow[j] = -1e30f; lrow[j] = 0.f; }

  for (int kt = 0; kt <= qt; ++kt) {
    // stage K tile (straight) and V tile (transposed)
#pragma unroll
    for (int c = 0; c < 4; ++c) {
      const int lin = tid + c * 128;
      const int r = lin >> 3, col = (lin & 7) << 3;
      *(u32x4*)(Ks + r * LD + col) = *(const u32x4*)(Kb + (size_t)(kt * 64 + r) * ld + col);
      union { u32x4 q; bf16_t e[8]; } vu;
      vu.q = *(const u32x4*)(Vb + (size_t)(kt * 64 + r) * ld + col);
#pragma unroll
      for (int e = 0; e < 8; ++e) Vt[(col + e) * LD + r] = vu.e[e];
    }
    __syncthreads();

    // S = Q K^T (16x64 per wave, 4 n-tiles, 2 K-steps of 32)
    v8f st[4];
#pragma unroll
    for (int nt = 0; nt < 4; ++nt) {
      v8f s = {};
#pragma unroll
      for (int ks = 0; ks < 2; ++ks)
        s = WMMA_BF16(qf[ks], frag_ld(Ks + (nt * 16 + nl) * LD + ks * 32, halfw), s);
      st[nt] = s;
    }

    // online softmax; row j+8*halfw lives in 16 lanes of this half-wave
#pragma unroll
    for (int j = 0; j < 8; ++j) {
      const int qi = qt * 64 + wid * 16 + halfw * 8 + j;
      float rm = -1e30f;
#pragma unroll
      for (int nt = 0; nt < 4; ++nt) {
        const int col = kt * 64 + nt * 16 + nl;
        const float v = (col <= qi) ? st[nt][j] * 0.125f : -1e30f;
        st[nt][j] = v;
        rm = fmaxf(rm, v);
      }
#pragma unroll
      for (int m = 1; m < 16; m <<= 1) rm = fmaxf(rm, __shfl_xor(rm, m, 32));
      const float mnew = fmaxf(mrow[j], rm);
      const float scale = __expf(mrow[j] - mnew);
      float rs = 0.f;
#pragma unroll
      for (int nt = 0; nt < 4; ++nt) {
        const float p = __expf(st[nt][j] - mnew);
        st[nt][j] = p;
        rs += p;
      }
#pragma unroll
      for (int m = 1; m < 16; m <<= 1) rs += __shfl_xor(rs, m, 32);
      lrow[j] = lrow[j] * scale + rs;
      mrow[j] = mnew;
#pragma unroll
      for (int d = 0; d < 4; ++d) oacc[d][j] *= scale;
    }

    // P: C layout -> LDS -> A layout
    bf16_t* pw = &Pb[wid][0];
#pragma unroll
    for (int nt = 0; nt < 4; ++nt)
#pragma unroll
      for (int j = 0; j < 8; ++j)
        pw[(halfw * 8 + j) * LD + nt * 16 + nl] = (bf16_t)st[nt][j];
    __syncthreads();   // guarantees DS store->load ordering

    v16bf pf[2];
#pragma unroll
    for (int ks = 0; ks < 2; ++ks) pf[ks] = frag_ld(pw + nl * LD + ks * 32, halfw);

    // O += P V  (keys are the K dim; Vt rows are output dims)
#pragma unroll
    for (int d = 0; d < 4; ++d)
#pragma unroll
      for (int ks = 0; ks < 2; ++ks)
        oacc[d] = WMMA_BF16(pf[ks], frag_ld(Vt + (d * 16 + nl) * LD + ks * 32, halfw), oacc[d]);

    __syncthreads();
  }

  // finalize: O / l
#pragma unroll
  for (int j = 0; j < 8; ++j) {
    const float inv = 1.f / lrow[j];
    const int row = qt * 64 + wid * 16 + halfw * 8 + j;
#pragma unroll
    for (int d = 0; d < 4; ++d)
      Ob[(size_t)row * ld + d * 16 + nl] = (bf16_t)(oacc[d][j] * inv);
  }
}

// ---------------- SiLU gate: u = silu(a) * g ----------------
__global__ void silu_mul_kernel(const bf16_t* __restrict__ a, const bf16_t* __restrict__ g,
                                bf16_t* __restrict__ out, long n) {
  long i = (long)blockIdx.x * blockDim.x + threadIdx.x;
  long stride = (long)gridDim.x * blockDim.x;
  for (; i < n; i += stride) {
    const float x = (float)a[i];
    const float y = (float)g[i];
    out[i] = (bf16_t)((x / (1.f + __expf(-x))) * y);
  }
}

// ---------------- host ----------------
extern "C" void kernel_launch(void* const* d_in, const int* in_sizes, int n_in,
                              void* d_out, int out_size, void* d_ws, size_t ws_size,
                              hipStream_t stream) {
  (void)in_sizes; (void)n_in; (void)out_size; (void)ws_size;
  const int B = 2, S = 2048, D = 1024, F = 4096, H = 16;
  const int M = B * S;

  const float* x  = (const float*)d_in[0];
  const float* wq = (const float*)d_in[1];
  const float* wk = (const float*)d_in[2];
  const float* wv = (const float*)d_in[3];
  const float* wo = (const float*)d_in[4];
  const float* w1 = (const float*)d_in[5];
  const float* w2 = (const float*)d_in[6];
  const float* w3 = (const float*)d_in[7];
  const float* g1 = (const float*)d_in[8];
  const float* g2 = (const float*)d_in[9];
  float* out = (float*)d_out;

  char* w = (char*)d_ws;
  auto alloc = [&](size_t bytes) -> char* {
    char* p = w; w += (bytes + 255) & ~(size_t)255; return p;
  };
  bf16_t* wq_b = (bf16_t*)alloc((size_t)D * D * 2);
  bf16_t* wk_b = (bf16_t*)alloc((size_t)D * D * 2);
  bf16_t* wv_b = (bf16_t*)alloc((size_t)D * D * 2);
  bf16_t* wo_b = (bf16_t*)alloc((size_t)D * D * 2);
  bf16_t* w1_b = (bf16_t*)alloc((size_t)F * D * 2);
  bf16_t* w3_b = (bf16_t*)alloc((size_t)F * D * 2);
  bf16_t* w2_b = (bf16_t*)alloc((size_t)D * F * 2);
  bf16_t* h1   = (bf16_t*)alloc((size_t)M * D * 2);
  bf16_t* qb   = (bf16_t*)alloc((size_t)M * D * 2);
  bf16_t* kb   = (bf16_t*)alloc((size_t)M * D * 2);
  bf16_t* vb   = (bf16_t*)alloc((size_t)M * D * 2);
  bf16_t* ob   = (bf16_t*)alloc((size_t)M * D * 2);
  float*  xmid = (float*) alloc((size_t)M * D * 4);
  bf16_t* h2   = (bf16_t*)alloc((size_t)M * D * 2);
  bf16_t* a1   = (bf16_t*)alloc((size_t)M * F * 2);
  bf16_t* a3   = (bf16_t*)alloc((size_t)M * F * 2);
  bf16_t* ug   = (bf16_t*)alloc((size_t)M * F * 2);

  // 1. cast weights
  cast_bf16_kernel<<<2048, 256, 0, stream>>>(wq, wq_b, (long)D * D);
  cast_bf16_kernel<<<2048, 256, 0, stream>>>(wk, wk_b, (long)D * D);
  cast_bf16_kernel<<<2048, 256, 0, stream>>>(wv, wv_b, (long)D * D);
  cast_bf16_kernel<<<2048, 256, 0, stream>>>(wo, wo_b, (long)D * D);
  cast_bf16_kernel<<<4096, 256, 0, stream>>>(w1, w1_b, (long)F * D);
  cast_bf16_kernel<<<4096, 256, 0, stream>>>(w3, w3_b, (long)F * D);
  cast_bf16_kernel<<<4096, 256, 0, stream>>>(w2, w2_b, (long)D * F);

  // 2. rmsnorm 1
  rmsnorm_kernel<<<M, 256, 0, stream>>>(x, g1, h1, D);

  // 3. q/k/v projections (NT)
  dim3 gD(D / GBN, M / GBM);
  gemm_bf16_nt_kernel<<<gD, 256, 0, stream>>>(h1, wq_b, nullptr, qb, nullptr, M, D, D);
  gemm_bf16_nt_kernel<<<gD, 256, 0, stream>>>(h1, wk_b, nullptr, kb, nullptr, M, D, D);
  gemm_bf16_nt_kernel<<<gD, 256, 0, stream>>>(h1, wv_b, nullptr, vb, nullptr, M, D, D);

  // 4. RoPE on q, k
  const long rtotal = (long)M * D / 2;
  rope_kernel<<<(unsigned)((rtotal + 255) / 256), 256, 0, stream>>>(qb, S, H, rtotal);
  rope_kernel<<<(unsigned)((rtotal + 255) / 256), 256, 0, stream>>>(kb, S, H, rtotal);

  // 5. causal flash attention
  attn_kernel<<<dim3(S / 64, B * H), 128, 0, stream>>>(qb, kb, vb, ob, S, H);

  // 6. output projection + residual -> xmid (f32)
  gemm_bf16_nt_kernel<<<gD, 256, 0, stream>>>(ob, wo_b, xmid, nullptr, x, M, D, D);

  // 7. rmsnorm 2
  rmsnorm_kernel<<<M, 256, 0, stream>>>(xmid, g2, h2, D);

  // 8. FFN up projections
  dim3 gF(F / GBN, M / GBM);
  gemm_bf16_nt_kernel<<<gF, 256, 0, stream>>>(h2, w1_b, nullptr, a1, nullptr, M, F, D);
  gemm_bf16_nt_kernel<<<gF, 256, 0, stream>>>(h2, w3_b, nullptr, a3, nullptr, M, F, D);

  // 9. SiLU gate
  silu_mul_kernel<<<8192, 256, 0, stream>>>(a1, a3, ug, (long)M * F);

  // 10. down projection + residual -> out (f32)
  gemm_bf16_nt_kernel<<<gD, 256, 0, stream>>>(ug, w2_b, out, nullptr, xmid, M, D, F);
}
